// GNN_43447889166647
// MI455X (gfx1250) — matrix operand order repaired and checked by reading, not compiled, not run on hardware
//
#include <hip/hip_runtime.h>
#include <hip/hip_bf16.h>

// ---------------------------------------------------------------------------
// GCN forward on gfx1250 (MI455X).
//   Layer: h = X @ W   (bf16 WMMA, fp32 accum; A and W pre-converted to bf16,
//                       W pre-transposed so B tiles are contiguous -> async LDS copy)
//          agg[d] = sum_e dinv[s]*ew*dinv[d]*h[s] + dinv[d]^2*h[d] + bias
//   L1: relu fused into bf16 conversion of layer-2 input; mean-pool; linear head.
// Inputs (setup_inputs order):
//   0 x[N,512] f32, 1 edge_index[2,E] i64, 2 batch[N] i64, 3 edge_col[E] f32,
//   4 W1[512,512] f32, 5 b1[512], 6 W2[512,512], 7 b2[512],
//   8 Wl[512,8], 9 bl[8].  Output: [G=64, 8] f32.
// ---------------------------------------------------------------------------

typedef __attribute__((ext_vector_type(16))) __bf16 v16bf;
typedef __attribute__((ext_vector_type(8)))  __bf16 v8bf;
typedef __attribute__((ext_vector_type(8)))  float  v8f;
typedef __attribute__((ext_vector_type(4)))  int    v4i;

#define F_IN 512
#define HDIM 512
#define KDIM 512           // inner dim of both GEMMs
#define OUTD 8
#define BLK_M 128
#define BLK_N 64

#if defined(__has_builtin)
# if __has_builtin(__builtin_amdgcn_global_load_async_to_lds_b128)
#  define HAVE_ASYNC_LDS 1
# endif
#endif

// Pointer types matching the async builtin's parameters:
//   (v4i addrspace(1)*, v4i addrspace(3)*, imm offset, imm cpol)
typedef __attribute__((address_space(1))) v4i* gv4ip;   // global
typedef __attribute__((address_space(3))) v4i* lv4ip;   // LDS

// Generic LDS pointers are {aperture | 32-bit LDS offset}; truncating to the
// low 32 bits yields the AS3 byte address.
__device__ __forceinline__ lv4ip to_lds_v4(void* p) {
    return (lv4ip)(size_t)(unsigned)(size_t)p;
}
__device__ __forceinline__ gv4ip to_glob_v4(const void* p) {
    return (gv4ip)(size_t)p;
}

__device__ __forceinline__ void atomic_add_f32(float* p, float v) {
    // relaxed, agent scope -> native global_atomic_add_f32 (no CAS loop)
    __hip_atomic_fetch_add(p, v, __ATOMIC_RELAXED, __HIP_MEMORY_SCOPE_AGENT);
}

// ---------------- degree / normalization --------------------------------
__global__ void deg_init_kernel(float* __restrict__ deg, int n) {
    int i = blockIdx.x * blockDim.x + threadIdx.x;
    if (i < n) deg[i] = 1.0f;           // self-loop weight
}

__global__ void deg_edge_kernel(const long long* __restrict__ dst,
                                const float* __restrict__ ew,
                                float* __restrict__ deg, int E) {
    int e = blockIdx.x * blockDim.x + threadIdx.x;
    if (e < E) atomic_add_f32(&deg[(int)dst[e]], ew[e]);
}

__global__ void dinv_kernel(float* __restrict__ deg, int n) {
    int i = blockIdx.x * blockDim.x + threadIdx.x;
    if (i < n) {
        float d = deg[i];
        deg[i] = d > 0.0f ? __frsqrt_rn(d) : 0.0f;   // in-place -> dinv
    }
}

// ---------------- precision conversion ----------------------------------
// fp32 -> bf16 elementwise (optionally fused relu); n is even.
__global__ void cvt_bf16_kernel(const float* __restrict__ in,
                                __bf16* __restrict__ out, size_t n, int do_relu) {
    size_t i = ((size_t)blockIdx.x * blockDim.x + threadIdx.x) * 2;
    if (i + 1 < n) {
        float2 v = *reinterpret_cast<const float2*>(in + i);
        if (do_relu) {
            v.x = v.x > 0.f ? v.x : 0.f;
            v.y = v.y > 0.f ? v.y : 0.f;
        }
        out[i]     = (__bf16)v.x;
        out[i + 1] = (__bf16)v.y;
    }
}

// W[K,Nc] fp32 -> WT[Nc,K] bf16 (transpose so B tiles are K-contiguous).
__global__ void cvt_transpose_bf16_kernel(const float* __restrict__ W,
                                          __bf16* __restrict__ WT, int Nc) {
    int idx = blockIdx.x * blockDim.x + threadIdx.x;    // KDIM * Nc work items
    if (idx >= KDIM * Nc) return;
    int k = idx & (KDIM - 1);
    int n = idx >> 9;                                   // KDIM == 512
    WT[(size_t)n * KDIM + k] = (__bf16)W[(size_t)k * Nc + n];
}

// ---------------- bf16 WMMA GEMM: C[M,Nc] = A[M,K] * W[K,Nc] ------------
// A: bf16 row-major [M,K]; WT: bf16 [Nc,K] (W transposed).
// block = 256 threads (8 waves); block tile 128(M) x 64(N); K%32==0.
__global__ __launch_bounds__(256)
void gemm_bf16_wmma(const __bf16* __restrict__ A, const __bf16* __restrict__ WT,
                    float* __restrict__ C, int M, int K, int Nc) {
    __shared__ __align__(128) __bf16 BT[BLK_N][32];   // [n][k] B slice, 4 KB

    const int tid  = threadIdx.x;
    const int wave = tid >> 5;
    const int lane = tid & 31;
    const int g    = lane >> 4;      // lane group 0/1
    const int ln   = lane & 15;      // 0..15
    const int n0   = blockIdx.y * BLK_N;
    const int rowBase = blockIdx.x * BLK_M + wave * 16;

    const int ar     = rowBase + ln;
    const int ar_c   = ar < M ? ar : (M - 1);          // clamp: tail rows unstored
    const __bf16* __restrict__ arow = A + (size_t)ar_c * K;

    // staging: thread t copies one 16-byte chunk per K-slice
    const int sn = tid >> 2;                 // 0..63  (row within N tile)
    const int sc = (tid & 3) * 8;            // 0,8,16,24 (bf16 offset in K slice)
    const __bf16* __restrict__ gsrc = WT + (size_t)(n0 + sn) * K + sc;
    __bf16* ldst = &BT[sn][sc];

    v8f acc0 = {0.f,0.f,0.f,0.f,0.f,0.f,0.f,0.f};
    v8f acc1 = acc0, acc2 = acc0, acc3 = acc0;

    for (int k0 = 0; k0 < K; k0 += 32) {
#if defined(HAVE_ASYNC_LDS)
        __builtin_amdgcn_global_load_async_to_lds_b128(
            to_glob_v4(gsrc + k0), to_lds_v4(ldst), 0, 0);
        asm volatile("s_wait_asynccnt 0x0" ::: "memory");
#else
        *reinterpret_cast<int4*>(ldst) =
            *reinterpret_cast<const int4*>(gsrc + k0);
#endif
        __syncthreads();

        // A fragment (contiguous k order, same as B):
        //   a[0..15] = A[m][k0+16g .. +15]
        v8bf lo = *reinterpret_cast<const v8bf*>(arow + k0 + 16 * g);
        v8bf hi = *reinterpret_cast<const v8bf*>(arow + k0 + 16 * g + 8);
        v16bf a = __builtin_shufflevector(lo, hi, 0,1,2,3,4,5,6,7,
                                                  8,9,10,11,12,13,14,15);

        // B fragments: col n = ln + 16*nt, K = 16g..16g+15 (32B contiguous LDS)
        v16bf b0 = *reinterpret_cast<const v16bf*>(&BT[ln][16 * g]);
        v16bf b1 = *reinterpret_cast<const v16bf*>(&BT[ln + 16][16 * g]);
        v16bf b2 = *reinterpret_cast<const v16bf*>(&BT[ln + 32][16 * g]);
        v16bf b3 = *reinterpret_cast<const v16bf*>(&BT[ln + 48][16 * g]);

        acc0 = __builtin_amdgcn_wmma_f32_16x16x32_bf16(false, a, false, b0,
                                                       (short)0, acc0, false, false);
        acc1 = __builtin_amdgcn_wmma_f32_16x16x32_bf16(false, a, false, b1,
                                                       (short)0, acc1, false, false);
        acc2 = __builtin_amdgcn_wmma_f32_16x16x32_bf16(false, a, false, b2,
                                                       (short)0, acc2, false, false);
        acc3 = __builtin_amdgcn_wmma_f32_16x16x32_bf16(false, a, false, b3,
                                                       (short)0, acc3, false, false);
        __syncthreads();
    }

    // C layout: VGPR r -> row rowBase + 8g + r, col = n0 + 16*nt + ln
    float* crow = C + (size_t)(rowBase + 8 * g) * Nc + n0 + ln;
    #pragma unroll
    for (int r = 0; r < 8; ++r) {
        int row = rowBase + 8 * g + r;
        if (row < M) {
            crow[0]  = acc0[r];
            crow[16] = acc1[r];
            crow[32] = acc2[r];
            crow[48] = acc3[r];
        }
        crow += Nc;
    }
}

// ---------------- aggregation: self loop + bias, then edge scatter ------
__global__ __launch_bounds__(256)
void agg_init_kernel(const float* __restrict__ h, const float* __restrict__ dinv,
                     const float* __restrict__ bias, float* __restrict__ agg,
                     int n) {
    int i = blockIdx.x;
    if (i >= n) return;
    float w = dinv[i] * dinv[i];                    // self-loop norm
    const float* hi = h + (size_t)i * HDIM;
    float* ai = agg + (size_t)i * HDIM;
    for (int f = threadIdx.x; f < HDIM; f += blockDim.x)
        ai[f] = w * hi[f] + bias[f];
}

__global__ __launch_bounds__(256)
void agg_edge_kernel(const long long* __restrict__ src,
                     const long long* __restrict__ dst,
                     const float* __restrict__ ew,
                     const float* __restrict__ dinv,
                     const float* __restrict__ h,
                     float* __restrict__ agg, int E) {
    int e = blockIdx.x;
    if (e >= E) return;
    int s = (int)src[e];
    int d = (int)dst[e];
    float norm = dinv[s] * ew[e] * dinv[d];
    const float* hs = h + (size_t)s * HDIM;
    float* ad = agg + (size_t)d * HDIM;
    for (int f = threadIdx.x; f < HDIM; f += blockDim.x)
        atomic_add_f32(&ad[f], norm * hs[f]);
}

// ---------------- pooling + head ----------------------------------------
__global__ void zero_kernel(float* __restrict__ p, int n) {
    int i = blockIdx.x * blockDim.x + threadIdx.x;
    if (i < n) p[i] = 0.0f;
}

__global__ __launch_bounds__(256)
void pool_kernel(const long long* __restrict__ batch,
                 const float* __restrict__ a,
                 float* __restrict__ pooled, float* __restrict__ cnts, int n) {
    int i = blockIdx.x;
    if (i >= n) return;
    int gi = (int)batch[i];
    const float* ai = a + (size_t)i * HDIM;
    float* pg = pooled + (size_t)gi * HDIM;
    for (int f = threadIdx.x; f < HDIM; f += blockDim.x)
        atomic_add_f32(&pg[f], ai[f]);
    if (threadIdx.x == 0) atomic_add_f32(&cnts[gi], 1.0f);
}

__global__ void final_kernel(const float* __restrict__ pooled,
                             const float* __restrict__ cnts,
                             const float* __restrict__ Wl,
                             const float* __restrict__ bl,
                             float* __restrict__ out, int G) {
    int idx = blockIdx.x * blockDim.x + threadIdx.x;
    if (idx >= G * OUTD) return;
    int g = idx / OUTD, o = idx % OUTD;
    float c = cnts[g];
    float inv = 1.0f / (c > 1.0f ? c : 1.0f);
    const float* pg = pooled + (size_t)g * HDIM;
    float sum = 0.0f;
    for (int k = 0; k < HDIM; ++k)
        sum += pg[k] * inv * Wl[k * OUTD + o];
    out[idx] = sum + bl[o];
}

// ---------------- launch ------------------------------------------------
extern "C" void kernel_launch(void* const* d_in, const int* in_sizes, int n_in,
                              void* d_out, int out_size, void* d_ws, size_t ws_size,
                              hipStream_t stream) {
    const float*      x     = (const float*)d_in[0];
    const long long*  ei    = (const long long*)d_in[1];   // int64 [2,E]
    const long long*  batch = (const long long*)d_in[2];   // int64 [N]
    const float*      ew    = (const float*)d_in[3];
    const float*      W1    = (const float*)d_in[4];
    const float*      b1    = (const float*)d_in[5];
    const float*      W2    = (const float*)d_in[6];
    const float*      b2    = (const float*)d_in[7];
    const float*      Wl    = (const float*)d_in[8];
    const float*      bl    = (const float*)d_in[9];
    float*            out   = (float*)d_out;

    const int N = in_sizes[0] / F_IN;
    const int E = in_sizes[3];
    const int G = out_size / OUTD;
    const long long* srcp = ei;
    const long long* dstp = ei + E;

    // workspace layout
    float* ws     = (float*)d_ws;
    float* dinv   = ws;                               // N f32 (deg -> dinv in place)
    float* buf0   = dinv + ((N + 15) & ~15);          // N*HDIM f32 (GEMM out)
    float* buf1   = buf0 + (size_t)N * HDIM;          // N*HDIM f32 (aggregated)
    float* pooled = buf1 + (size_t)N * HDIM;          // G*HDIM f32
    float* cnts   = pooled + (size_t)G * HDIM;        // G f32
    __bf16* Abf   = (__bf16*)(cnts + ((G + 15) & ~15));  // N*KDIM bf16
    __bf16* WbfT  = Abf + (size_t)N * KDIM;              // HDIM*KDIM bf16

    const int T = 256;
    const size_t NH = (size_t)N * HDIM;
    dim3 gemm_grid((N + BLK_M - 1) / BLK_M, HDIM / BLK_N);

    // 1) normalization coefficients
    deg_init_kernel<<<(N + T - 1) / T, T, 0, stream>>>(dinv, N);
    deg_edge_kernel<<<(E + T - 1) / T, T, 0, stream>>>(dstp, ew, dinv, E);
    dinv_kernel<<<(N + T - 1) / T, T, 0, stream>>>(dinv, N);

    // 2) layer 1: h = x@W1 ; agg
    cvt_bf16_kernel<<<(int)((NH / 2 + T - 1) / T), T, 0, stream>>>(x, Abf, NH, 0);
    cvt_transpose_bf16_kernel<<<(KDIM * HDIM + T - 1) / T, T, 0, stream>>>(W1, WbfT, HDIM);
    gemm_bf16_wmma<<<gemm_grid, T, 0, stream>>>(Abf, WbfT, buf0, N, KDIM, HDIM);
    agg_init_kernel<<<N, T, 0, stream>>>(buf0, dinv, b1, buf1, N);
    agg_edge_kernel<<<E, T, 0, stream>>>(srcp, dstp, ew, dinv, buf0, buf1, E);

    // 3) layer 2: h = relu(agg1)@W2 ; agg   (relu fused into bf16 conversion)
    cvt_bf16_kernel<<<(int)((NH / 2 + T - 1) / T), T, 0, stream>>>(buf1, Abf, NH, 1);
    cvt_transpose_bf16_kernel<<<(KDIM * HDIM + T - 1) / T, T, 0, stream>>>(W2, WbfT, HDIM);
    gemm_bf16_wmma<<<gemm_grid, T, 0, stream>>>(Abf, WbfT, buf0, N, KDIM, HDIM);
    agg_init_kernel<<<N, T, 0, stream>>>(buf0, dinv, b2, buf1, N);
    agg_edge_kernel<<<E, T, 0, stream>>>(srcp, dstp, ew, dinv, buf0, buf1, E);

    // 4) mean-pool + linear head
    zero_kernel<<<(G * HDIM + G + T - 1) / T, T, 0, stream>>>(pooled, G * HDIM + G);
    pool_kernel<<<N, T, 0, stream>>>(batch, buf1, pooled, cnts, N);
    final_kernel<<<(G * OUTD + T - 1) / T, T, 0, stream>>>(pooled, cnts, Wl, bl, out, G);
}
